// AttnUpBlock2D_60799557042830
// MI455X (gfx1250) — matrix operand order repaired and checked by reading, not compiled, not run on hardware
//
#include <hip/hip_runtime.h>
#include <math.h>

// ---------------------------------------------------------------------------
// AttnUpBlock2D on gfx1250 (MI455X) — all GEMMs on V_WMMA_F32_16X16X32_F16
// (16384 FLOP/instr, f32 accumulate), f16 operand staging, aligned wide loads.
//   B=4, Cin=256, Cout=128, H=W=64, N=4096/batch, head dims: qk=64, v=128
// Conv K-dim is reordered as k' = dx*(CIN*3) + (ci*3+dy) so each 32-wide
// K-chunk has a wave-uniform dx shift -> B operands are gathered per lane as 16
// scalar LDS loads from patch rows at column lq + dx.
// ---------------------------------------------------------------------------

typedef __attribute__((ext_vector_type(16))) _Float16 v16h;
typedef __attribute__((ext_vector_type(8)))  _Float16 v8h;
typedef __attribute__((ext_vector_type(8)))  float    v8f;

#define BN_EPS 1e-5f
#define SLOPE  0.2f

__device__ __forceinline__ v8f wmma16(v16h a, v16h b, v8f c) {
  return __builtin_amdgcn_wmma_f32_16x16x32_f16(
      /*neg_a=*/false, a, /*neg_b=*/false, b,
      /*c_mod=*/(short)0, c, /*reuse_a=*/false, /*reuse_b=*/false);
}

__device__ __forceinline__ float lrelu(float x) {
  return x >= 0.0f ? x : SLOPE * x;
}

// Build a v16h A-operand from two aligned 8-half runs (ISA 16-bit A layout:
// lane half hf: e<8 -> K=8*hf+e ; e>=8 -> K=16+8*hf+(e-8)).
__device__ __forceinline__ v16h load_a16(const _Float16* run0, const _Float16* run1) {
  v8h lo = *(const v8h*)run0;
  v8h hi = *(const v8h*)run1;
  v16h a;
#pragma unroll
  for (int e = 0; e < 8; ++e) { a[e] = lo[e]; a[8 + e] = hi[e]; }
  return a;
}

// ---------------------------------------------------------------------------
// conv weight convert + K-reorder:  out[co][dx*(CIN*3) + ci*3 + dy] = W[co][ci][dy][dx]
// ---------------------------------------------------------------------------
__global__ void reorder_w_kernel(const float* __restrict__ in,
                                 _Float16* __restrict__ out, int CIN) {
  int K = CIN * 9;
  int t = blockIdx.x * blockDim.x + threadIdx.x;
  if (t >= 128 * K) return;
  int co = t / K;
  int kp = t - co * K;
  int dx = kp / (CIN * 3);
  int rc = kp - dx * (CIN * 3);
  int ci = rc / 3;
  int dy = rc - ci * 3;
  out[t] = (_Float16)in[((co * CIN + ci) * 3 + dy) * 3 + dx];
}

// ---------------------------------------------------------------------------
// 1) bilinear upsample 32->64 (align_corners), emit f16.  4*256*64*64 elems.
// ---------------------------------------------------------------------------
__global__ void upsample2x_kernel(const float* __restrict__ in,
                                  _Float16* __restrict__ out) {
  int t = blockIdx.x * blockDim.x + threadIdx.x;
  if (t >= 4 * 256 * 64 * 64) return;
  int xo = t & 63;
  int yo = (t >> 6) & 63;
  int c  = (t >> 12) & 255;
  int b  = t >> 20;
  const float sc = 31.0f / 63.0f;
  float fy = yo * sc, fx = xo * sc;
  int y0 = (int)fy, x0 = (int)fx;
  int y1 = y0 + 1 > 31 ? 31 : y0 + 1;
  int x1 = x0 + 1 > 31 ? 31 : x0 + 1;
  float wy = fy - (float)y0, wx = fx - (float)x0;
  const float* p = in + ((b * 256 + c) << 10);
  float a00 = p[y0 * 32 + x0], a01 = p[y0 * 32 + x1];
  float a10 = p[y1 * 32 + x0], a11 = p[y1 * 32 + x1];
  float r0 = a00 * (1.0f - wy) + a10 * wy;
  float r1 = a01 * (1.0f - wy) + a11 * wy;
  out[t] = (_Float16)(r0 * (1.0f - wx) + r1 * wx);
}

// ---------------------------------------------------------------------------
// 2-4) implicit-GEMM 3x3 conv + (bias) + BN + lrelu.
//   8 waves / workgroup; workgroup = one 16-pixel strip x all 128 co.
//   LDS: zero-padded halo patch [ci*3+dy][24] f16 (stride padded 18->24 for
//   16B-aligned rows), shared by the 8 waves.
// ---------------------------------------------------------------------------
template <int CIN>
__global__ void __launch_bounds__(256)
conv3x3_f16_kernel(const _Float16* __restrict__ in_h,   // [B, CIN, 64, 64]
                   const _Float16* __restrict__ w_h,    // [128, CIN*9] (reordered)
                   const float* __restrict__ bias, int has_bias,
                   const float* __restrict__ gam,
                   const float* __restrict__ bet,
                   const float* __restrict__ rmean,
                   const float* __restrict__ rvar,
                   float* __restrict__ out_f32,         // [B, 128, 64, 64]
                   _Float16* __restrict__ out_f16, int write_f16) {
  __shared__ __align__(32) _Float16 patch[CIN * 72];    // [rc = ci*3+dy][24]
  const int K  = CIN * 9;
  const int KC = CIN * 3;                               // rows per dx block
  int tid = threadIdx.x;
  int strip = blockIdx.x;            // (b, y, x0/16)
  int xt = (strip & 3) << 4;
  int y  = (strip >> 2) & 63;
  int b  = strip >> 8;
  const _Float16* inb = in_h + b * CIN * 4096;

  // cooperative halo fill (zero-padded; xx in [0,18) valid, [18,24) zero pad)
  for (int idx = tid; idx < CIN * 72; idx += 256) {
    int rc = idx / 24;
    int xx = idx - rc * 24;
    int ci = rc / 3;
    int dy = rc - ci * 3;
    int iy = y + dy - 1;
    int ix = xt + xx - 1;
    _Float16 v = (_Float16)0.0f;
    if (xx < 18 && iy >= 0 && iy < 64 && ix >= 0 && ix < 64)
      v = inb[(ci << 12) + (iy << 6) + ix];
    patch[idx] = v;
  }
  __syncthreads();

  int wave = tid >> 5;
  int lane = tid & 31;
  int lq = lane & 15, hf = lane >> 4;
  int m0 = wave << 4;                               // 16 output channels / wave
  const _Float16* arow = w_h + (m0 + lq) * K;

  v8f acc = {};
  for (int kk = 0; kk < K; kk += 32) {
    // A: weight chunk, two aligned 8-half runs (already K-reordered)
    v16h a = load_a16(arow + kk + (hf << 3), arow + kk + 16 + (hf << 3));
    // B: column n = lq (pixel), rows rc = (kk % KC) + k in table order, wave-uniform shift dx = kk / KC
    int dx = kk / KC;                               // uniform over the wave
    int rb = kk - dx * KC;
    v16h bm;
#pragma unroll
    for (int e = 0; e < 16; ++e) {
      int k = (e < 8) ? ((hf << 3) + e) : (16 + (hf << 3) + (e - 8));
      bm[e] = patch[(rb + k) * 24 + lq + dx];
    }
    acc = wmma16(a, bm, acc);
  }

  // epilogue: (+bias)*bn_scale + bn_shift, lrelu.  C layout: row=r+8*hf, col=lq
  int sp = (y << 6) + xt + lq;
#pragma unroll
  for (int r = 0; r < 8; ++r) {
    int co = m0 + r + (hf << 3);
    float scale = gam[co] * rsqrtf(rvar[co] + BN_EPS);
    float shift = bet[co] - rmean[co] * scale;
    float val = acc[r];
    if (has_bias) val += bias[co];
    val = lrelu(val * scale + shift);
    int idx = (b * 128 + co) * 4096 + sp;
    out_f32[idx] = val;
    if (write_f16) out_f16[idx] = (_Float16)val;
  }
}

// ---------------------------------------------------------------------------
// 5) projections (f32 math, f16 stores in WMMA-friendly layouts)
//    q -> [b, n, 64]   k -> [b, 64, n]   v -> [b, n, 128] (transposed!)
// ---------------------------------------------------------------------------
__global__ void proj_q_kernel(const float* __restrict__ xin,
                              const float* __restrict__ W,
                              const float* __restrict__ bias,
                              _Float16* __restrict__ qout) {
  int t = blockIdx.x * blockDim.x + threadIdx.x;
  if (t >= 4 * 4096 * 64) return;
  int o = t & 63;
  int nn = (t >> 6) & 4095;
  int b = t >> 18;
  const float* xb = xin + b * 128 * 4096 + nn;
  const float* wr = W + o * 128;
  float acc = bias[o];
  for (int c = 0; c < 128; ++c) acc += wr[c] * xb[c << 12];
  qout[t] = (_Float16)acc;             // [b][n][o]
}

__global__ void proj_k_kernel(const float* __restrict__ xin,
                              const float* __restrict__ W,
                              const float* __restrict__ bias,
                              _Float16* __restrict__ kout) {
  int t = blockIdx.x * blockDim.x + threadIdx.x;
  if (t >= 4 * 64 * 4096) return;
  int nn = t & 4095;
  int o = (t >> 12) & 63;
  int b = t >> 18;
  const float* xb = xin + b * 128 * 4096 + nn;
  const float* wr = W + o * 128;
  float acc = bias[o];
  for (int c = 0; c < 128; ++c) acc += wr[c] * xb[c << 12];
  kout[t] = (_Float16)acc;             // [b][o][n]
}

__global__ void proj_vT_kernel(const float* __restrict__ xin,
                               const float* __restrict__ W,
                               const float* __restrict__ bias,
                               _Float16* __restrict__ vout) {
  int t = blockIdx.x * blockDim.x + threadIdx.x;
  if (t >= 4 * 4096 * 128) return;
  int o = t & 127;
  int nn = (t >> 7) & 4095;
  int b = t >> 19;
  const float* xb = xin + b * 128 * 4096 + nn;
  const float* wr = W + o * 128;
  float acc = bias[o];
  for (int c = 0; c < 128; ++c) acc += wr[c] * xb[c << 12];
  vout[t] = (_Float16)acc;             // [b][n][o]  (V transposed)
}

// ---------------------------------------------------------------------------
// 6) flash attention: one wave = 16 query rows; 32 keys / iteration.
//    S(16x32): 4 wmma (K=64);  online softmax;  P -> f16 via 1KB LDS;
//    O(16x128) += P(16x32) VT: 8 wmma.  Epilogue: O/l + y2 + y0, lrelu.
// ---------------------------------------------------------------------------
__global__ void __launch_bounds__(32)
flash_attn_fused_kernel(const _Float16* __restrict__ q,   // [b, n, 64]
                        const _Float16* __restrict__ kmat,// [b, 64, n]
                        const _Float16* __restrict__ vT,  // [b, n, 128]
                        const float* __restrict__ y2,
                        const float* __restrict__ y0,
                        float* __restrict__ out) {
  __shared__ __align__(32) _Float16 pbuf[16 * 32];
  int lane = threadIdx.x;
  int lq = lane & 15, hf = lane >> 4;
  int b  = blockIdx.x >> 8;
  int m0 = (blockIdx.x & 255) << 4;

  // Q block (A layout), two K-chunks of 32
  const _Float16* qrow = q + ((b << 12) + m0 + lq) * 64;
  v16h qa[2];
#pragma unroll
  for (int kk = 0; kk < 2; ++kk)
    qa[kk] = load_a16(qrow + kk * 32 + (hf << 3),
                      qrow + kk * 32 + 16 + (hf << 3));

  const _Float16* kb  = kmat + b * 64 * 4096;
  const _Float16* vtb = vT + (b << 12) * 128;

  v8f o_acc[8];
#pragma unroll
  for (int g = 0; g < 8; ++g) o_acc[g] = (v8f){};
  float rm[8], rl[8];
#pragma unroll
  for (int r = 0; r < 8; ++r) { rm[r] = -1e30f; rl[r] = 0.0f; }

  for (int n0 = 0; n0 < 4096; n0 += 32) {
    // ---- S = Q * K_tile : two 16x16 f32 tiles ----
    v8f s0 = {}, s1 = {};
#pragma unroll
    for (int kk = 0; kk < 2; ++kk) {
      v16h b0, b1;
#pragma unroll
      for (int e = 0; e < 16; ++e) {
        int k = (e < 8) ? ((hf << 3) + e) : (16 + (hf << 3) + (e - 8));
        const _Float16* krow = kb + (kk * 32 + k) * 4096;
        b0[e] = krow[n0 + lq];
        b1[e] = krow[n0 + 16 + lq];
      }
      s0 = wmma16(qa[kk], b0, s0);
      s1 = wmma16(qa[kk], b1, s1);
    }
    // ---- online softmax over 32 columns (rows live in 16-lane halves) ----
    float alpha[8];
#pragma unroll
    for (int r = 0; r < 8; ++r) {
      float tm = fmaxf(s0[r], s1[r]);
#pragma unroll
      for (int off = 1; off < 16; off <<= 1)
        tm = fmaxf(tm, __shfl_xor(tm, off, 32));
      float nm = fmaxf(rm[r], tm);
      alpha[r] = __expf(rm[r] - nm);
      rm[r] = nm;
      float p0 = __expf(s0[r] - nm);
      float p1 = __expf(s1[r] - nm);
      s0[r] = p0; s1[r] = p1;
      float ts = p0 + p1;
#pragma unroll
      for (int off = 1; off < 16; off <<= 1)
        ts += __shfl_xor(ts, off, 32);
      rl[r] = rl[r] * alpha[r] + ts;
    }
#pragma unroll
    for (int g = 0; g < 8; ++g)
#pragma unroll
      for (int r = 0; r < 8; ++r) o_acc[g][r] *= alpha[r];

    // ---- P (16x32) -> f16 A-matrix via LDS ----
    __syncthreads();
#pragma unroll
    for (int r = 0; r < 8; ++r) {
      int m = r + (hf << 3);
      pbuf[m * 32 + lq]      = (_Float16)s0[r];
      pbuf[m * 32 + 16 + lq] = (_Float16)s1[r];
    }
    __syncthreads();
    v16h pa = load_a16(pbuf + lq * 32 + (hf << 3),
                       pbuf + lq * 32 + 16 + (hf << 3));
    // ---- O += P * VT_tile : B col c = g*16+lq, rows n = n0+k (table order) ----
#pragma unroll
    for (int g = 0; g < 8; ++g) {
      v16h bv;
#pragma unroll
      for (int e = 0; e < 16; ++e) {
        int k = (e < 8) ? ((hf << 3) + e) : (16 + (hf << 3) + (e - 8));
        bv[e] = vtb[(n0 + k) * 128 + (g << 4) + lq];
      }
      o_acc[g] = wmma16(pa, bv, o_acc[g]);
    }
    __syncthreads();
  }

  // ---- epilogue: normalize, + attn residual (y2) + identity (y0), lrelu ----
#pragma unroll
  for (int g = 0; g < 8; ++g) {
    int c = (g << 4) + lq;
#pragma unroll
    for (int r = 0; r < 8; ++r) {
      int m = m0 + r + (hf << 3);
      int idx = ((b << 7) + c) * 4096 + m;
      float val = o_acc[g][r] / rl[r];
      val += y2[idx] + y0[idx];
      out[idx] = lrelu(val);
    }
  }
}

// ---------------------------------------------------------------------------
// launcher
// ---------------------------------------------------------------------------
extern "C" void kernel_launch(void* const* d_in, const int* in_sizes, int n_in,
                              void* d_out, int out_size, void* d_ws, size_t ws_size,
                              hipStream_t stream) {
  const float* x    = (const float*)d_in[0];
  const float* W_up = (const float*)d_in[1];
  const float* b_up = (const float*)d_in[2];
  const float* g0   = (const float*)d_in[3];
  const float* be0  = (const float*)d_in[4];
  const float* m0   = (const float*)d_in[5];
  const float* v0   = (const float*)d_in[6];
  const float* W_r0 = (const float*)d_in[7];
  const float* g1   = (const float*)d_in[8];
  const float* be1  = (const float*)d_in[9];
  const float* m1   = (const float*)d_in[10];
  const float* v1   = (const float*)d_in[11];
  const float* W_r1 = (const float*)d_in[12];
  const float* g2   = (const float*)d_in[13];
  const float* be2  = (const float*)d_in[14];
  const float* m2   = (const float*)d_in[15];
  const float* v2   = (const float*)d_in[16];
  const float* Wq   = (const float*)d_in[17];
  const float* bq   = (const float*)d_in[18];
  const float* Wk   = (const float*)d_in[19];
  const float* bk   = (const float*)d_in[20];
  const float* Wv   = (const float*)d_in[21];
  const float* bv   = (const float*)d_in[22];
  float* out = (float*)d_out;

  // ---- workspace layout ----
  float* wsf = (float*)d_ws;
  float* y0f = wsf;                    // 4*128*4096 = 2097152 f32
  float* y1f = y0f + 2097152;
  float* y2f = y1f + 2097152;
  _Float16* wsh  = (_Float16*)(y2f + 2097152);
  _Float16* xu_h = wsh;                // 4*256*4096 = 4194304
  _Float16* y0h  = xu_h + 4194304;     // 2097152
  _Float16* y1h  = y0h + 2097152;      // 2097152
  _Float16* wuph = y1h + 2097152;      // 128*2304 = 294912
  _Float16* wr0h = wuph + 294912;      // 128*1152 = 147456
  _Float16* wr1h = wr0h + 147456;      // 147456
  _Float16* qh   = wr1h + 147456;      // 4*4096*64 = 1048576
  _Float16* kh   = qh + 1048576;       // 1048576
  _Float16* vth  = kh + 1048576;       // 4*4096*128 = 2097152

  // ---- weight conversion + K-reorder to f16 ----
  reorder_w_kernel<<<(294912 + 255) / 256, 256, 0, stream>>>(W_up, wuph, 256);
  reorder_w_kernel<<<(147456 + 255) / 256, 256, 0, stream>>>(W_r0, wr0h, 128);
  reorder_w_kernel<<<(147456 + 255) / 256, 256, 0, stream>>>(W_r1, wr1h, 128);

  // 1) upsample -> f16
  upsample2x_kernel<<<(4 * 256 * 64 * 64) / 256, 256, 0, stream>>>(x, xu_h);
  // 2) conv1 + bias + BN0 + lrelu -> y0 (f32 identity + f16 feed)
  conv3x3_f16_kernel<256><<<4096, 256, 0, stream>>>(
      xu_h, wuph, b_up, 1, g0, be0, m0, v0, y0f, y0h, 1);
  // 3) conv2 + BN1 + lrelu -> y1 (f16 feed)
  conv3x3_f16_kernel<128><<<4096, 256, 0, stream>>>(
      y0h, wr0h, b_up, 0, g1, be1, m1, v1, y1f, y1h, 1);
  // 4) conv3 + BN2 + lrelu -> y2 (f32 for residual + projections)
  conv3x3_f16_kernel<128><<<4096, 256, 0, stream>>>(
      y1h, wr1h, b_up, 0, g2, be2, m2, v2, y2f, y1h, 0);
  // 5) q/k/v projections (f16 outputs in WMMA layouts)
  proj_q_kernel<<<(4 * 4096 * 64) / 256, 256, 0, stream>>>(y2f, Wq, bq, qh);
  proj_k_kernel<<<(4 * 64 * 4096) / 256, 256, 0, stream>>>(y2f, Wk, bk, kh);
  proj_vT_kernel<<<(4 * 4096 * 128) / 256, 256, 0, stream>>>(y2f, Wv, bv, vth);
  // 6) flash attention + residuals + lrelu -> out
  flash_attn_fused_kernel<<<1024, 32, 0, stream>>>(qh, kh, vth, y2f, y0f, out);
}
